// SpliceMamba_21388937134869
// MI455X (gfx1250) — matrix operand order repaired and checked by reading, not compiled, not run on hardware
//
#include <hip/hip_runtime.h>
#include <cstdint>
#include <cstddef>

// ---------------- types ----------------
typedef __bf16 bf16_t;
typedef __attribute__((ext_vector_type(16))) __bf16 v16bf;
typedef __attribute__((ext_vector_type(8)))  __bf16 v8bf;
typedef __attribute__((ext_vector_type(8)))  float  v8f;
typedef __attribute__((ext_vector_type(4)))  int    v4i;

// ---------------- model constants ----------------
#define BATCH    2
#define SEQ      2048
#define TTOK     (BATCH*SEQ)        // 4096 tokens
#define DMODEL   256
#define DINNER   512
#define DSTATE   64
#define DCONV    4
#define HEADD    32
#define NH       16                  // mamba heads
#define DINPROJ  1160                // 2*512 + 2*64 + 16
#define CONVDIM  640                 // 512 + 128
#define NLAYERS  8
#define NATTN    4
#define NHEADS   8
#define BLKA     512
#define WRAD     400
#define CHUNKQ   128
#define NCHUNK   (TTOK/CHUNKQ)       // 32 (chunks never straddle batches)
#define EPSF     1e-5f

static inline int cdiv(int a, int b) { return (a + b - 1) / b; }

// optional CDNA5 async global->LDS path (probe via __has_builtin; falls back cleanly)
#if defined(__has_builtin)
#  if __has_builtin(__builtin_amdgcn_global_load_async_to_lds_b128) && \
      __has_builtin(__builtin_amdgcn_s_wait_asynccnt)
#    define GEMM_ASYNC 1
#  endif
#endif
#ifndef GEMM_ASYNC
#  define GEMM_ASYNC 0
#endif

// ---------------- device math helpers ----------------
__device__ __forceinline__ float geluf(float x) {
    return 0.5f * x * (1.f + erff(x * 0.70710678118654752f));
}
__device__ __forceinline__ float siluf(float x) {
    return x / (1.f + expf(-x));
}
__device__ __forceinline__ float softplusf(float x) {
    return (x > 20.f) ? x : log1pf(expf(x));
}

// single consistent dynamic-LDS symbol across all kernels
extern __shared__ float smF[];

// =====================================================================
// WMMA GEMM:  C[M,N] = act( A[M,K](bf16) * W[N,K]^T(bf16) + bias )
// block = 256 threads = 8 waves. Block tile: 128 rows x 64 cols.
// Each wave: 16 rows x 64 cols = 4 accumulators; A-frag loaded once per
// 32-K step and reused across 4 B-tiles. All 4 B-frags are preloaded
// before the 4 WMMAs so DS waits overlap with matrix math.
// 64 x K weight tile staged in LDS (async global->LDS when available).
// =====================================================================
__global__ void k_gemm_bf16(const bf16_t* __restrict__ A, const bf16_t* __restrict__ W,
                            const float* __restrict__ bias, float* __restrict__ C,
                            int M, int N, int K, int act) {
    bf16_t* sW = (bf16_t*)smF;                 // 64 * K bf16
    const int lane = threadIdx.x & 31;
    const int wave = threadIdx.x >> 5;
    const int colBase = blockIdx.x * 64;
    const int rowBase = (blockIdx.y * 8 + wave) * 16;

    // cooperative stage of the 64 x K weight tile (zero-fill past N)
    const int kvecs = K >> 3;                  // K is always a multiple of 32
    for (int idx = threadIdx.x; idx < 64 * kvecs; idx += blockDim.x) {
        int n  = idx / kvecs;
        int kc = (idx % kvecs) * 8;
        if (colBase + n < N) {
#if GEMM_ASYNC
            __builtin_amdgcn_global_load_async_to_lds_b128(
                (v4i*)(W + (size_t)(colBase + n) * K + kc),
                (v4i*)(sW + n * K + kc),
                0, 0);
#else
            *(v8bf*)(sW + n * K + kc) = *(const v8bf*)(W + (size_t)(colBase + n) * K + kc);
#endif
        } else {
            v8bf z;
            #pragma unroll
            for (int i = 0; i < 8; ++i) z[i] = (bf16_t)0.f;
            *(v8bf*)(sW + n * K + kc) = z;
        }
    }
#if GEMM_ASYNC
    __builtin_amdgcn_s_wait_asynccnt(0);
#endif
    __syncthreads();
    if (rowBase >= M) return;

    const int m   = lane & 15;
    const int row = rowBase + m;
    const int kbA = (lane >> 4) * 8;           // A-frag K base within 32
    const int nB  = lane & 15;
    const int kbB = (lane >> 4) * 8;           // B-frag K base within 32

    v8f acc[4];
    #pragma unroll
    for (int c = 0; c < 4; ++c) acc[c] = (v8f){};

    const bf16_t* arow = A + (size_t)row * K;
    for (int k0 = 0; k0 < K; k0 += 32) {
        const bf16_t* ap = arow + k0 + kbA;
        __builtin_prefetch((const void*)(ap + 64), 0, 3);
        v8bf alo = *(const v8bf*)ap;
        v8bf ahi = *(const v8bf*)(ap + 16);
        // preload all 4 B fragments before issuing any WMMA
        v8bf blo[4], bhi[4];
        #pragma unroll
        for (int c = 0; c < 4; ++c) {
            const bf16_t* bp = sW + (c * 16 + nB) * K + k0 + kbB;
            blo[c] = *(const v8bf*)bp;
            bhi[c] = *(const v8bf*)(bp + 16);
        }
        v16bf a;
        #pragma unroll
        for (int i = 0; i < 8; ++i) { a[i] = alo[i]; a[8 + i] = ahi[i]; }
        #pragma unroll
        for (int c = 0; c < 4; ++c) {
            v16bf b;
            #pragma unroll
            for (int i = 0; i < 8; ++i) { b[i] = blo[c][i]; b[8 + i] = bhi[c][i]; }
            acc[c] = __builtin_amdgcn_wmma_f32_16x16x32_bf16(
                false, a, false, b, (short)0, acc[c], false, false);
        }
    }

    const int rowOut = rowBase + (lane >> 4) * 8;
    #pragma unroll
    for (int c = 0; c < 4; ++c) {
        int col = colBase + c * 16 + (lane & 15);
        if (col < N) {
            float bv = bias ? bias[col] : 0.f;
            #pragma unroll
            for (int r = 0; r < 8; ++r) {
                int rr = rowOut + r;
                if (rr < M) {
                    float v = acc[c][r] + bv;
                    if (act == 1) v = geluf(v);
                    C[(size_t)rr * N + col] = v;
                }
            }
        }
    }
}

// ---------------- elementwise / conversion kernels ----------------
__global__ void k_tobf16(const float* __restrict__ in, bf16_t* __restrict__ out, int n) {
    int i = blockIdx.x * blockDim.x + threadIdx.x;
    if (i < n) out[i] = (bf16_t)in[i];
}
__global__ void k_cat_bf16(const float* __restrict__ a, const float* __restrict__ b,
                           bf16_t* __restrict__ out) {  // [T,512] = cat(hf,hb)
    int i = blockIdx.x * blockDim.x + threadIdx.x;
    if (i >= TTOK * 512) return;
    int c = i & 511, t = i >> 9;
    float v = (c < DMODEL) ? a[(size_t)t * DMODEL + c] : b[(size_t)t * DMODEL + (c - DMODEL)];
    out[i] = (bf16_t)v;
}
__global__ void k_add(float* __restrict__ dst, const float* __restrict__ src, int n) {
    int i = blockIdx.x * blockDim.x + threadIdx.x;
    if (i < n) dst[i] += src[i];
}
__global__ void k_copy(float* __restrict__ dst, const float* __restrict__ src, int n) {
    int i = blockIdx.x * blockDim.x + threadIdx.x;
    if (i < n) dst[i] = src[i];
}
__global__ void k_flip(const float* __restrict__ src, float* __restrict__ dst) {
    int i = blockIdx.x * blockDim.x + threadIdx.x;
    if (i >= TTOK * DMODEL) return;
    int c = i & (DMODEL - 1);
    int tok = i >> 8;
    int b = tok / SEQ, t = tok % SEQ;
    dst[i] = src[((size_t)(b * SEQ + (SEQ - 1 - t))) * DMODEL + c];
}

// ---------------- stem ----------------
__global__ void k_stem(const float* __restrict__ x,
                       const float* w5, const float* b5, const float* g5, const float* be5,
                       const float* w11, const float* b11, const float* g11, const float* be11,
                       float* __restrict__ hcat) {  // token-major [T,256]
    int idx = blockIdx.x * blockDim.x + threadIdx.x;
    if (idx >= TTOK * DMODEL) return;
    int c = idx & (DMODEL - 1);
    int tok = idx >> 8;
    int b = tok / SEQ, t = tok % SEQ;
    float rs = rsqrtf(1.f + EPSF);
    float a;
    if (c < 128) {
        a = b5[c];
        for (int ci = 0; ci < 4; ++ci)
            for (int k = 0; k < 5; ++k) {
                int tt = t + k - 2;
                if (tt >= 0 && tt < SEQ)
                    a += x[((size_t)b * 4 + ci) * SEQ + tt] * w5[(c * 4 + ci) * 5 + k];
            }
        a = a * (g5[c] * rs) + be5[c];
    } else {
        int c2 = c - 128;
        a = b11[c2];
        for (int ci = 0; ci < 4; ++ci)
            for (int k = 0; k < 11; ++k) {
                int tt = t + k - 5;
                if (tt >= 0 && tt < SEQ)
                    a += x[((size_t)b * 4 + ci) * SEQ + tt] * w11[(c2 * 4 + ci) * 11 + k];
            }
        a = a * (g11[c2] * rs) + be11[c2];
    }
    hcat[idx] = geluf(a);
}

__global__ void k_bn_gelu_pos(const float* __restrict__ in, const float* g,
                              const float* be, float* __restrict__ out) {
    int idx = blockIdx.x * blockDim.x + threadIdx.x;
    if (idx >= TTOK * DMODEL) return;
    int c = idx & (DMODEL - 1);
    int tok = idx >> 8;
    int t = tok % SEQ;
    float rs = rsqrtf(1.f + EPSF);
    float v = geluf(in[idx] * (g[c] * rs) + be[c]);
    int j2 = c & ~1;
    float div = expf(-(float)j2 * (9.210340371976184f / 256.f));  // ln(10000)/d
    float ang = (float)t * div;
    v += (c & 1) ? cosf(ang) : sinf(ang);
    out[idx] = v;
}

// ---------------- norms ----------------
__global__ void k_layernorm(const float* __restrict__ in, const float* __restrict__ w,
                            const float* __restrict__ b, float* __restrict__ out,
                            int T_, int D) {
    int t = blockIdx.x * blockDim.x + threadIdx.x;
    if (t >= T_) return;
    const float* p = in + (size_t)t * D;
    float m = 0.f;
    for (int d = 0; d < D; ++d) m += p[d];
    m /= (float)D;
    float v = 0.f;
    for (int d = 0; d < D; ++d) { float q = p[d] - m; v += q * q; }
    float rs = rsqrtf(v / (float)D + EPSF);
    float* o = out + (size_t)t * D;
    for (int d = 0; d < D; ++d) o[d] = (p[d] - m) * rs * w[d] + b[d];
}

__global__ void k_gate_rms(float* __restrict__ y, const float* __restrict__ zx,
                           const float* __restrict__ nw) {
    int t = blockIdx.x * blockDim.x + threadIdx.x;
    if (t >= TTOK) return;
    size_t yb = (size_t)t * DINNER, zb = (size_t)t * DINPROJ;
    float ss = 0.f;
    for (int d = 0; d < DINNER; ++d) {
        float g = y[yb + d] * siluf(zx[zb + d]);
        y[yb + d] = g;
        ss += g * g;
    }
    float rs = rsqrtf(ss / (float)DINNER + EPSF);
    for (int d = 0; d < DINNER; ++d) y[yb + d] *= rs * nw[d];
}

// ---------------- mamba pieces ----------------
__global__ void k_dwconv_silu(const float* __restrict__ zx, const float* __restrict__ cw,
                              const float* __restrict__ cb, float* __restrict__ xBCa) {
    int idx = blockIdx.x * blockDim.x + threadIdx.x;
    if (idx >= TTOK * CONVDIM) return;
    int c = idx % CONVDIM;
    int tok = idx / CONVDIM;
    int b = tok / SEQ, t = tok % SEQ;
    float a = cb[c];
    #pragma unroll
    for (int k = 0; k < DCONV; ++k) {
        int tt = t - (DCONV - 1) + k;
        if (tt >= 0)
            a += zx[((size_t)(b * SEQ + tt)) * DINPROJ + DINNER + c] * cw[c * DCONV + k];
    }
    xBCa[(size_t)tok * CONVDIM + c] = siluf(a);
}

__global__ void k_dt_xdt(const float* __restrict__ zx, const float* __restrict__ xBCa,
                         const float* __restrict__ dtb, const float* __restrict__ Alog,
                         float* __restrict__ xdt, float* __restrict__ dA) {
    int idx = blockIdx.x * blockDim.x + threadIdx.x;
    if (idx >= TTOK * NH) return;
    int h = idx & (NH - 1);
    int tok = idx >> 4;
    float dt = softplusf(zx[(size_t)tok * DINPROJ + DINNER + CONVDIM + h] + dtb[h]);
    dA[idx] = dt * (-expf(Alog[h]));
    size_t xb = (size_t)tok * CONVDIM + h * HEADD;
    size_t ob = (size_t)tok * DINNER + h * HEADD;
    #pragma unroll
    for (int p = 0; p < HEADD; ++p) xdt[ob + p] = xBCa[xb + p] * dt;
}

__global__ void k_cumsum(const float* __restrict__ dA, float* __restrict__ cs) {
    int idx = blockIdx.x * blockDim.x + threadIdx.x;
    if (idx >= NCHUNK * NH) return;
    int h = idx & (NH - 1);
    int gc = idx >> 4;
    float a = 0.f;
    for (int q = 0; q < CHUNKQ; ++q) {
        a += dA[((size_t)gc * CHUNKQ + q) * NH + h];
        cs[(size_t)idx * CHUNKQ + q] = a;
    }
}

// intra-chunk (Yd) : block = (chunk, head), 128 threads (thread = query i)
__global__ void k_intra(const float* __restrict__ xBCa, const float* __restrict__ xdt,
                        const float* __restrict__ cs, float* __restrict__ y) {
    float* sB = smF;              // 128*64
    float* sC = sB + 8192;        // 128*64
    float* sX = sC + 8192;        // 128*32
    float* sS = sX + 4096;        // 128
    int gc = blockIdx.x / NH, h = blockIdx.x % NH;
    int tid = threadIdx.x;
    for (int i = tid; i < 8192; i += 128) {
        int j = i >> 6, n = i & 63;
        size_t rb = ((size_t)gc * CHUNKQ + j) * CONVDIM + DINNER;
        sB[i] = xBCa[rb + n];
        sC[i] = xBCa[rb + DSTATE + n];
    }
    for (int i = tid; i < 4096; i += 128) {
        int j = i >> 5, p = i & 31;
        sX[i] = xdt[((size_t)gc * CHUNKQ + j) * DINNER + h * HEADD + p];
    }
    sS[tid] = cs[((size_t)gc * NH + h) * CHUNKQ + tid];
    __syncthreads();
    int i = tid;
    float csi = sS[i];
    float acc[HEADD];
    #pragma unroll
    for (int p = 0; p < HEADD; ++p) acc[p] = 0.f;
    for (int j = 0; j <= i; ++j) {
        float s = 0.f;
        for (int n = 0; n < DSTATE; ++n) s += sC[i * 64 + n] * sB[j * 64 + n];
        float w = expf(csi - sS[j]) * s;
        #pragma unroll
        for (int p = 0; p < HEADD; ++p) acc[p] += w * sX[j * 32 + p];
    }
    size_t tok = (size_t)gc * CHUNKQ + i;
    #pragma unroll
    for (int p = 0; p < HEADD; ++p) y[tok * DINNER + h * HEADD + p] = acc[p];
}

// per-chunk states : block = (chunk, head), 256 threads (thread -> 8 (p,n) pairs)
__global__ void k_states(const float* __restrict__ xBCa, const float* __restrict__ xdt,
                         const float* __restrict__ cs, float* __restrict__ states,
                         float* __restrict__ cdec) {
    float* sB = smF;              // 128*64
    float* sX = sB + 8192;        // 128*32
    float* sW = sX + 4096;        // 128
    int gc = blockIdx.x / NH, h = blockIdx.x % NH;
    int tid = threadIdx.x;
    for (int i = tid; i < 8192; i += 256) {
        int j = i >> 6, n = i & 63;
        sB[i] = xBCa[((size_t)gc * CHUNKQ + j) * CONVDIM + DINNER + n];
    }
    for (int i = tid; i < 4096; i += 256) {
        int j = i >> 5, p = i & 31;
        sX[i] = xdt[((size_t)gc * CHUNKQ + j) * DINNER + h * HEADD + p];
    }
    const float* csp = cs + ((size_t)gc * NH + h) * CHUNKQ;
    float csl = csp[CHUNKQ - 1];
    if (tid < 128) sW[tid] = expf(csl - csp[tid]);
    if (tid == 0) cdec[gc * NH + h] = expf(csl);
    __syncthreads();
    size_t base = ((size_t)gc * NH + h) * 2048;
    #pragma unroll
    for (int r = 0; r < 8; ++r) {
        int pn = tid * 8 + r;
        int p = pn >> 6, n = pn & 63;
        float a = 0.f;
        for (int j = 0; j < CHUNKQ; ++j) a += sB[j * 64 + n] * sW[j] * sX[j * 32 + p];
        states[base + pn] = a;
    }
}

// sequential inter-chunk scan : block = (batch, head)
__global__ void k_scan(const float* __restrict__ states, const float* __restrict__ cdec,
                       float* __restrict__ prev) {
    int b = blockIdx.x / NH, h = blockIdx.x % NH;
    int tid = threadIdx.x;  // 256
    float carry[8];
    #pragma unroll
    for (int i = 0; i < 8; ++i) carry[i] = 0.f;
    const int ncl = SEQ / CHUNKQ;  // 16
    for (int c = 0; c < ncl; ++c) {
        int gc = b * ncl + c;
        size_t base = ((size_t)gc * NH + h) * 2048;
        float dec = cdec[gc * NH + h];
        #pragma unroll
        for (int i = 0; i < 8; ++i) {
            int e = tid * 8 + i;
            prev[base + e] = carry[i];
            carry[i] = carry[i] * dec + states[base + e];
        }
    }
}

// inter-chunk contribution (Yo) + D*x skip : block = (chunk, head), 128 threads
__global__ void k_inter(const float* __restrict__ xBCa, const float* __restrict__ cs,
                        const float* __restrict__ prev, const float* __restrict__ Dp,
                        float* __restrict__ y) {
    float* sP = smF;              // 2048
    float* sC = sP + 2048;        // 128*64
    float* sS = sC + 8192;        // 128
    int gc = blockIdx.x / NH, h = blockIdx.x % NH;
    int tid = threadIdx.x;
    size_t pbase = ((size_t)gc * NH + h) * 2048;
    for (int i = tid; i < 2048; i += 128) sP[i] = prev[pbase + i];
    for (int i = tid; i < 8192; i += 128) {
        int j = i >> 6, n = i & 63;
        sC[i] = xBCa[((size_t)gc * CHUNKQ + j) * CONVDIM + DINNER + DSTATE + n];
    }
    sS[tid] = cs[((size_t)gc * NH + h) * CHUNKQ + tid];
    __syncthreads();
    int i = tid;
    float e = expf(sS[i]);
    float Dh = Dp[h];
    size_t tok = (size_t)gc * CHUNKQ + i;
    #pragma unroll
    for (int p = 0; p < HEADD; ++p) {
        float a = 0.f;
        for (int n = 0; n < DSTATE; ++n) a += sC[i * 64 + n] * sP[p * 64 + n];
        float xs = xBCa[tok * CONVDIM + h * HEADD + p];
        y[tok * DINNER + h * HEADD + p] += e * a + Dh * xs;
    }
}

// ---------------- windowed attention (online softmax) ----------------
__global__ void k_attn(const float* __restrict__ q, const float* __restrict__ k,
                       const float* __restrict__ v, float* __restrict__ o) {
    int idx = blockIdx.x * blockDim.x + threadIdx.x;
    if (idx >= BATCH * 4 * NHEADS * BLKA) return;
    int qi = idx & (BLKA - 1);
    int h  = (idx >> 9) & (NHEADS - 1);
    int g  = (idx >> 12) & 3;
    int b  = idx >> 14;
    int tok = b * SEQ + g * BLKA + qi;
    float qv[HEADD];
    #pragma unroll
    for (int d = 0; d < HEADD; ++d) qv[d] = q[(size_t)tok * DMODEL + h * HEADD + d];
    float m = -3.0e38f, l = 0.f, acc[HEADD];
    #pragma unroll
    for (int d = 0; d < HEADD; ++d) acc[d] = 0.f;
    int jlo = qi + (BLKA - WRAD);
    int jhi = qi + (BLKA + WRAD);
    for (int j = jlo; j <= jhi; ++j) {
        int kj = (g - 1) * BLKA + j;
        if (kj < 0 || kj >= SEQ) continue;
        int kt = b * SEQ + kj;
        const float* kp = k + (size_t)kt * DMODEL + h * HEADD;
        const float* vp = v + (size_t)kt * DMODEL + h * HEADD;
        float s = 0.f;
        #pragma unroll
        for (int d = 0; d < HEADD; ++d) s += qv[d] * kp[d];
        s *= 0.17677669529663688f;  // 1/sqrt(32)
        float mn = fmaxf(m, s);
        float sc = expf(m - mn);
        float pr = expf(s - mn);
        l = l * sc + pr;
        #pragma unroll
        for (int d = 0; d < HEADD; ++d) acc[d] = acc[d] * sc + pr * vp[d];
        m = mn;
    }
    float inv = 1.f / l;
    #pragma unroll
    for (int d = 0; d < HEADD; ++d) o[(size_t)tok * DMODEL + h * HEADD + d] = acc[d] * inv;
}

// ---------------- classifier head (N=3, naive) ----------------
__global__ void k_head_out(const float* __restrict__ h1, const float* __restrict__ w2,
                           const float* __restrict__ b2, float* __restrict__ out) {
    int idx = blockIdx.x * blockDim.x + threadIdx.x;
    if (idx >= TTOK * 3) return;
    int n = idx % 3, t = idx / 3;
    float a = b2[n];
    const float* p = h1 + (size_t)t * DMODEL;
    const float* w = w2 + n * DMODEL;
    for (int d = 0; d < DMODEL; ++d) a += p[d] * w[d];
    out[idx] = a;
}

// =====================================================================
// host-side orchestration
// =====================================================================
struct Scratch {
    float *work, *hn, *tmp, *zx, *xBCa, *xdt, *dA, *cs, *states, *prev, *cdec, *y;
    bf16_t *bfA, *bfW;
};

static void gemm(hipStream_t st, const bf16_t* A, const bf16_t* W, const float* bias,
                 float* C, int M, int N, int K, int act) {
    dim3 g(cdiv(N, 64), cdiv(M, 128));
    k_gemm_bf16<<<g, 256, (size_t)64 * K * sizeof(bf16_t), st>>>(A, W, bias, C, M, N, K, act);
}

static void gemm_ff(hipStream_t st, Scratch& S, const float* A, const float* W,
                    const float* bias, float* C, int M, int N, int K, int act) {
    k_tobf16<<<cdiv(M * K, 256), 256, 0, st>>>(A, S.bfA, M * K);
    k_tobf16<<<cdiv(N * K, 256), 256, 0, st>>>(W, S.bfW, N * K);
    gemm(st, S.bfA, S.bfW, bias, C, M, N, K, act);
}

static void mamba_stack(hipStream_t st, Scratch& S,
                        const float* lnw, const float* lnb, const float* inp,
                        const float* cw, const float* cb, const float* dtb,
                        const float* Al, const float* Dp, const float* nw,
                        const float* op) {
    for (int l = 0; l < NLAYERS; ++l) {
        k_layernorm<<<cdiv(TTOK, 256), 256, 0, st>>>(S.work, lnw + l * DMODEL,
                                                     lnb + l * DMODEL, S.hn, TTOK, DMODEL);
        k_tobf16<<<cdiv(TTOK * DMODEL, 256), 256, 0, st>>>(S.hn, S.bfA, TTOK * DMODEL);
        k_tobf16<<<cdiv(DINPROJ * DMODEL, 256), 256, 0, st>>>(
            inp + (size_t)l * DINPROJ * DMODEL, S.bfW, DINPROJ * DMODEL);
        gemm(st, S.bfA, S.bfW, nullptr, S.zx, TTOK, DINPROJ, DMODEL, 0);

        k_dwconv_silu<<<cdiv(TTOK * CONVDIM, 256), 256, 0, st>>>(
            S.zx, cw + (size_t)l * CONVDIM * DCONV, cb + (size_t)l * CONVDIM, S.xBCa);
        k_dt_xdt<<<cdiv(TTOK * NH, 256), 256, 0, st>>>(S.zx, S.xBCa, dtb + l * NH,
                                                       Al + l * NH, S.xdt, S.dA);
        k_cumsum<<<cdiv(NCHUNK * NH, 256), 256, 0, st>>>(S.dA, S.cs);
        k_intra<<<NCHUNK * NH, 128, 20608 * sizeof(float), st>>>(S.xBCa, S.xdt, S.cs, S.y);
        k_states<<<NCHUNK * NH, 256, 12416 * sizeof(float), st>>>(S.xBCa, S.xdt, S.cs,
                                                                  S.states, S.cdec);
        k_scan<<<BATCH * NH, 256, 0, st>>>(S.states, S.cdec, S.prev);
        k_inter<<<NCHUNK * NH, 128, 10368 * sizeof(float), st>>>(S.xBCa, S.cs, S.prev,
                                                                 Dp + l * NH, S.y);
        k_gate_rms<<<cdiv(TTOK, 256), 256, 0, st>>>(S.y, S.zx, nw + (size_t)l * DINNER);

        k_tobf16<<<cdiv(TTOK * DINNER, 256), 256, 0, st>>>(S.y, S.bfA, TTOK * DINNER);
        k_tobf16<<<cdiv(DMODEL * DINNER, 256), 256, 0, st>>>(
            op + (size_t)l * DMODEL * DINNER, S.bfW, DMODEL * DINNER);
        gemm(st, S.bfA, S.bfW, nullptr, S.tmp, TTOK, DMODEL, DINNER, 0);
        k_add<<<cdiv(TTOK * DMODEL, 256), 256, 0, st>>>(S.work, S.tmp, TTOK * DMODEL);
    }
}

extern "C" void kernel_launch(void* const* d_in, const int* in_sizes, int n_in,
                              void* d_out, int out_size, void* d_ws, size_t ws_size,
                              hipStream_t stream) {
    (void)in_sizes; (void)n_in; (void)out_size; (void)ws_size;
    auto IN = [&](int i) { return (const float*)d_in[i]; };

    const float* X     = IN(0);
    const float* s_w5  = IN(1),  *s_b5  = IN(2),  *s_g5  = IN(3),  *s_be5 = IN(4);
    const float* s_w11 = IN(5),  *s_b11 = IN(6),  *s_g11 = IN(7),  *s_be11= IN(8);
    const float* s_wp  = IN(9),  *s_bp  = IN(10), *s_gp  = IN(11), *s_bep = IN(12);
    const float* f_lnw = IN(13), *f_lnb = IN(14), *f_inp = IN(15), *f_cw  = IN(16);
    const float* f_cb  = IN(17), *f_dtb = IN(18), *f_Al  = IN(19), *f_D   = IN(20);
    const float* f_nw  = IN(21), *f_op  = IN(22);
    const float* b_lnw = IN(23), *b_lnb = IN(24), *b_inp = IN(25), *b_cw  = IN(26);
    const float* b_cb  = IN(27), *b_dtb = IN(28), *b_Al  = IN(29), *b_D   = IN(30);
    const float* b_nw  = IN(31), *b_op  = IN(32);
    const float* fusion= IN(33);
    const float* c_w1  = IN(34), *c_b1  = IN(35), *c_w2  = IN(36), *c_b2  = IN(37);
    const float* r_w1  = IN(38), *r_b1  = IN(39), *r_w2  = IN(40), *r_b2  = IN(41);
    const float* a_lnw = IN(42), *a_lnb = IN(43), *a_wq  = IN(44), *a_bq  = IN(45);
    const float* a_wk  = IN(46), *a_bk  = IN(47), *a_wv  = IN(48), *a_bv  = IN(49);
    const float* a_wo  = IN(50), *a_bo  = IN(51);

    // bump allocator over workspace (256B aligned slabs)
    char* base = (char*)d_ws;
    size_t off = 0;
    auto AF = [&](size_t elems) -> float* {
        float* p = (float*)(base + off);
        off += ((elems * sizeof(float) + 255) & ~(size_t)255);
        return p;
    };
    auto AB = [&](size_t elems) -> bf16_t* {
        bf16_t* p = (bf16_t*)(base + off);
        off += ((elems * sizeof(bf16_t) + 255) & ~(size_t)255);
        return p;
    };

    float* horig = AF((size_t)TTOK * DMODEL);
    float* hf    = AF((size_t)TTOK * DMODEL);
    float* hb    = AF((size_t)TTOK * DMODEL);
    float* enc   = AF((size_t)TTOK * DMODEL);
    Scratch S;
    S.work   = AF((size_t)TTOK * DMODEL);
    S.hn     = AF((size_t)TTOK * DMODEL);
    S.tmp    = AF((size_t)TTOK * DMODEL);
    S.zx     = AF((size_t)TTOK * DINPROJ);
    S.xBCa   = AF((size_t)TTOK * CONVDIM);
    S.xdt    = AF((size_t)TTOK * DINNER);
    S.dA     = AF((size_t)TTOK * NH);
    S.cs     = AF((size_t)NCHUNK * NH * CHUNKQ);
    S.states = AF((size_t)NCHUNK * NH * 2048);
    S.prev   = AF((size_t)NCHUNK * NH * 2048);
    S.cdec   = AF((size_t)NCHUNK * NH);
    S.y      = AF((size_t)TTOK * DINNER);
    S.bfA    = AB((size_t)TTOK * DINNER);
    S.bfW    = AB((size_t)DINPROJ * DMODEL);
    float* qb = AF((size_t)TTOK * DMODEL);
    float* kb = AF((size_t)TTOK * DMODEL);
    float* vb = AF((size_t)TTOK * DMODEL);
    float* ob = AF((size_t)TTOK * DMODEL);
    float* h1 = AF((size_t)TTOK * DMODEL);

    // ---------------- stem ----------------
    float* hcat = S.tmp;
    k_stem<<<cdiv(TTOK * DMODEL, 256), 256, 0, stream>>>(
        X, s_w5, s_b5, s_g5, s_be5, s_w11, s_b11, s_g11, s_be11, hcat);
    gemm_ff(stream, S, hcat, s_wp, s_bp, S.work, TTOK, DMODEL, DMODEL, 0);
    k_bn_gelu_pos<<<cdiv(TTOK * DMODEL, 256), 256, 0, stream>>>(S.work, s_gp, s_bep, horig);

    // ---------------- forward mamba stack ----------------
    k_copy<<<cdiv(TTOK * DMODEL, 256), 256, 0, stream>>>(S.work, horig, TTOK * DMODEL);
    mamba_stack(stream, S, f_lnw, f_lnb, f_inp, f_cw, f_cb, f_dtb, f_Al, f_D, f_nw, f_op);
    k_copy<<<cdiv(TTOK * DMODEL, 256), 256, 0, stream>>>(hf, S.work, TTOK * DMODEL);

    // ---------------- backward mamba stack (on flipped sequence) ----------------
    k_flip<<<cdiv(TTOK * DMODEL, 256), 256, 0, stream>>>(horig, S.work);
    mamba_stack(stream, S, b_lnw, b_lnb, b_inp, b_cw, b_cb, b_dtb, b_Al, b_D, b_nw, b_op);
    k_flip<<<cdiv(TTOK * DMODEL, 256), 256, 0, stream>>>(S.work, hb);

    // ---------------- fusion ----------------
    k_cat_bf16<<<cdiv(TTOK * 512, 256), 256, 0, stream>>>(hf, hb, S.bfA);
    k_tobf16<<<cdiv(DMODEL * 512, 256), 256, 0, stream>>>(fusion, S.bfW, DMODEL * 512);
    gemm(stream, S.bfA, S.bfW, nullptr, enc, TTOK, DMODEL, 512, 0);

    // ---------------- coarse head ----------------
    gemm_ff(stream, S, enc, c_w1, c_b1, h1, TTOK, DMODEL, DMODEL, 1);
    k_head_out<<<cdiv(TTOK * 3, 256), 256, 0, stream>>>(h1, c_w2, c_b2, (float*)d_out);

    // ---------------- SWA attention stack ----------------
    k_copy<<<cdiv(TTOK * DMODEL, 256), 256, 0, stream>>>(S.work, enc, TTOK * DMODEL);
    for (int al = 0; al < NATTN; ++al) {
        const size_t wstride = (size_t)DMODEL * DMODEL;
        k_layernorm<<<cdiv(TTOK, 256), 256, 0, stream>>>(
            S.work, a_lnw + al * DMODEL, a_lnb + al * DMODEL, S.hn, TTOK, DMODEL);
        k_tobf16<<<cdiv(TTOK * DMODEL, 256), 256, 0, stream>>>(S.hn, S.bfA, TTOK * DMODEL);
        k_tobf16<<<cdiv((int)wstride, 256), 256, 0, stream>>>(a_wq + al * wstride, S.bfW, (int)wstride);
        gemm(stream, S.bfA, S.bfW, a_bq + al * DMODEL, qb, TTOK, DMODEL, DMODEL, 0);
        k_tobf16<<<cdiv((int)wstride, 256), 256, 0, stream>>>(a_wk + al * wstride, S.bfW, (int)wstride);
        gemm(stream, S.bfA, S.bfW, a_bk + al * DMODEL, kb, TTOK, DMODEL, DMODEL, 0);
        k_tobf16<<<cdiv((int)wstride, 256), 256, 0, stream>>>(a_wv + al * wstride, S.bfW, (int)wstride);
        gemm(stream, S.bfA, S.bfW, a_bv + al * DMODEL, vb, TTOK, DMODEL, DMODEL, 0);
        k_attn<<<cdiv(BATCH * 4 * NHEADS * BLKA, 256), 256, 0, stream>>>(qb, kb, vb, ob);
        k_tobf16<<<cdiv(TTOK * DMODEL, 256), 256, 0, stream>>>(ob, S.bfA, TTOK * DMODEL);
        k_tobf16<<<cdiv((int)wstride, 256), 256, 0, stream>>>(a_wo + al * wstride, S.bfW, (int)wstride);
        gemm(stream, S.bfA, S.bfW, a_bo + al * DMODEL, S.tmp, TTOK, DMODEL, DMODEL, 0);
        k_add<<<cdiv(TTOK * DMODEL, 256), 256, 0, stream>>>(S.work, S.tmp, TTOK * DMODEL);
    }

    // ---------------- refined head + enc output ----------------
    gemm_ff(stream, S, S.work, r_w1, r_b1, h1, TTOK, DMODEL, DMODEL, 1);
    k_head_out<<<cdiv(TTOK * 3, 256), 256, 0, stream>>>(h1, r_w2, r_b2,
                                                        (float*)d_out + (size_t)TTOK * 3);
    k_copy<<<cdiv(TTOK * DMODEL, 256), 256, 0, stream>>>(
        (float*)d_out + (size_t)2 * TTOK * 3, enc, TTOK * DMODEL);
}
